// coAttention_42726334661172
// MI455X (gfx1250) — matrix operand order repaired and checked by reading, not compiled, not run on hardware
//
#include <hip/hip_runtime.h>

// ---------------------------------------------------------------------------
// Live computation after DCE (B=32, D3=DQ=1024):
//   q[b]   = Wp @ Q[b] + bp[i]
//   out[b] = relu(W3 @ M[b] + W4 @ q[b] + b3[i]) * q[b]
// Inputs: 0:K(dead) 1:Q 2:M 3:ht(dead) 4:Wp 5:bp 6:W3 7:W4 8:b3 9:Wd 10:bd
//
// Strategy: bf16 hi/lo split GEMMs (3 WMMAs per product, fp32-class accuracy)
// on v_wmma_f32_16x16x32_bf16. Weights pre-converted once to bf16 hi/lo
// planes in d_ws so the hot loop has no fp32->bf16 VALU work. Fused per
// (batch, 32-col k-strip): q panel lives in LDS (hi/lo), never hits HBM.
// ---------------------------------------------------------------------------

#define DD   1024      // matrix dimension
#define NB   32        // batch
#define KT   32        // output columns per workgroup (k-strip)
#define JC   32        // reduction chunk (WMMA bf16 K)
#define SROW (JC + 8)  // staged tile row length (ushorts), padded
#define QROW (DD + 8)  // q-panel row length (ushorts), padded

typedef __attribute__((ext_vector_type(16))) __bf16         v16bf;
typedef __attribute__((ext_vector_type(8)))  float          v8f;
typedef __attribute__((ext_vector_type(4)))  float          f32x4;
typedef __attribute__((ext_vector_type(2)))  float          f32x2;
typedef __attribute__((ext_vector_type(4)))  unsigned int   u32x4;
typedef __attribute__((ext_vector_type(4)))  unsigned short u16x4;

union V16 {
    v16bf          v;
    unsigned short us[16];
    u32x4          q[2];
};

// float -> bf16 (RNE) and back, via bit math only.
__device__ __forceinline__ unsigned short f2bf(float f) {
    unsigned u = __float_as_uint(f);
    u += 0x7FFFu + ((u >> 16) & 1u);
    return (unsigned short)(u >> 16);
}
__device__ __forceinline__ float bf2f(unsigned short h) {
    return __uint_as_float(((unsigned)h) << 16);
}

// A operand (16x32 bf16, MxK) from a row-major bf16 plane:
// lane (m = lane%16, h = lane/16); element e: K = e + 16h
//   -> elements 0..7  = plane[row*DD + j0 + 16h     .. +7]  (one 16B load)
//   -> elements 8..15 = plane[row*DD + j0 + 16h + 8 .. +7]  (one 16B load)
__device__ __forceinline__ void loadA(const unsigned short* __restrict__ hiP,
                                      const unsigned short* __restrict__ loP,
                                      size_t rowOff, int h, V16& hi, V16& lo) {
    const unsigned short* ph = hiP + rowOff + 16 * h;
    const unsigned short* pl = loP + rowOff + 16 * h;
    hi.q[0] = *(const u32x4*)(ph);
    hi.q[1] = *(const u32x4*)(ph + 8);
    lo.q[0] = *(const u32x4*)(pl);
    lo.q[1] = *(const u32x4*)(pl + 8);
}

// B operand (32x16 bf16, KxN) from an LDS plane stored [n-col][K-contig]:
// lane (n = lane%16, h = lane/16); element e: K = e + 16h -> one contiguous
// 16-ushort (32B) run at plane[row*stride + col0 + 16h].
__device__ __forceinline__ void loadB(const unsigned short* __restrict__ plane,
                                      int stride, int row, int col, V16& v) {
    const u32x4* p = (const u32x4*)(plane + row * stride + col);
    v.q[0] = p[0];
    v.q[1] = p[1];
}

__device__ __forceinline__ v8f wmma_bf16(const V16& a, const V16& b, v8f c) {
    // (neg_a, A, neg_b, B, c_mod, C, reuse_a, reuse_b)
    return __builtin_amdgcn_wmma_f32_16x16x32_bf16(
        false, a.v, false, b.v, (short)0, c, false, false);
}

// ---------------------------------------------------------------------------
// Pre-pass: convert Wp/W3/W4 (fp32, row-major) to bf16 hi/lo planes in ws.
// ---------------------------------------------------------------------------
__global__ __launch_bounds__(256)
void weight_prep_kernel(const float* __restrict__ Wp, const float* __restrict__ W3,
                        const float* __restrict__ W4,
                        unsigned short* __restrict__ ws) {
    const size_t n   = (size_t)DD * DD;
    const size_t idx = ((size_t)blockIdx.x * blockDim.x + threadIdx.x) * 4;
    const float* src[3]     = {Wp, W3, W4};
#pragma unroll
    for (int m = 0; m < 3; ++m) {
        f32x4 f = *(const f32x4*)(src[m] + idx);
        u16x4 hi, lo;
#pragma unroll
        for (int e = 0; e < 4; ++e) {
            unsigned short h = f2bf(f[e]);
            hi[e] = h;
            lo[e] = f2bf(f[e] - bf2f(h));
        }
        *(u16x4*)(ws + (size_t)(2 * m + 0) * n + idx) = hi;
        *(u16x4*)(ws + (size_t)(2 * m + 1) * n + idx) = lo;
    }
}

// ---------------------------------------------------------------------------
// Fused main kernel: one workgroup per (batch, 32-column k-strip).
// ---------------------------------------------------------------------------
__global__ __launch_bounds__(512)
void coattn_fused_kernel(const float* __restrict__ Q, const float* __restrict__ M,
                         const unsigned short* __restrict__ WpHi,
                         const unsigned short* __restrict__ WpLo,
                         const unsigned short* __restrict__ W3Hi,
                         const unsigned short* __restrict__ W3Lo,
                         const unsigned short* __restrict__ W4Hi,
                         const unsigned short* __restrict__ W4Lo,
                         const float* __restrict__ bp, const float* __restrict__ b3,
                         float* __restrict__ out) {
    // q panel for this k-strip: hi/lo bf16, layout [k_local][j] (K contiguous)
    __shared__ unsigned short qhi[KT * QROW];   // 32*1032*2B = 64.5 KB
    __shared__ unsigned short qlo[KT * QROW];
    // staged data tile (Q or M chunk), transposed: [k_local][j]
    __shared__ unsigned short shi[KT * SROW];
    __shared__ unsigned short slo[KT * SROW];

    const int blk  = blockIdx.x;
    const int b    = blk / (DD / KT);
    const int k0   = (blk % (DD / KT)) * KT;
    const int tid  = threadIdx.x;
    const int lane = tid & 31;
    const int wave = tid >> 5;          // 0..15
    const int h    = lane >> 4;         // half-wave: 0 or 1
    const int ln   = lane & 15;

    const float* Qb = Q + (size_t)b * DD * DD;
    const float* Mb = M + (size_t)b * DD * DD;
    const int rowBase = wave * 64;      // 4 row-tiles of 16 per wave

    const v8f vzero = {0.f, 0.f, 0.f, 0.f, 0.f, 0.f, 0.f, 0.f};
    v8f acc[4][2];

    // ---------------- Phase 1: q panel = Wp @ Q[:, k-strip] + bp -----------
#pragma unroll
    for (int t = 0; t < 4; ++t)
#pragma unroll
        for (int nb = 0; nb < 2; ++nb) acc[t][nb] = vzero;

    for (int jc = 0; jc < DD / JC; ++jc) {
        const int j0 = jc * JC;
        __syncthreads();
        {   // stage Q[j0..j0+31][k0..k0+31] transposed into shi/slo
            const int p = tid * 2;            // 512 thr * 2 = 1024 elems
            const int j = p >> 5;
            const int k = p & 31;
            f32x2 f = *(const f32x2*)(Qb + (size_t)(j0 + j) * DD + k0 + k);
            unsigned short h0 = f2bf(f.x);
            shi[(k + 0) * SROW + j] = h0;
            slo[(k + 0) * SROW + j] = f2bf(f.x - bf2f(h0));
            unsigned short h1 = f2bf(f.y);
            shi[(k + 1) * SROW + j] = h1;
            slo[(k + 1) * SROW + j] = f2bf(f.y - bf2f(h1));
        }
        __syncthreads();
        // B fragments are shared across all row-tiles of this wave: load once
        V16 bhi[2], blo[2];
#pragma unroll
        for (int nb = 0; nb < 2; ++nb) {
            loadB(shi, SROW, nb * 16 + ln, 16 * h, bhi[nb]);
            loadB(slo, SROW, nb * 16 + ln, 16 * h, blo[nb]);
        }
#pragma unroll
        for (int t = 0; t < 4; ++t) {
            const int i0 = rowBase + t * 16;
            V16 ahi, alo;
            loadA(WpHi, WpLo, (size_t)(i0 + ln) * DD + j0, h, ahi, alo);
#pragma unroll
            for (int nb = 0; nb < 2; ++nb) {
                v8f c = acc[t][nb];
                c = wmma_bf16(ahi, bhi[nb], c);
                c = wmma_bf16(ahi, blo[nb], c);
                c = wmma_bf16(alo, bhi[nb], c);
                acc[t][nb] = c;
            }
        }
    }

    // write q (+bp row bias) into LDS planes as hi/lo bf16
#pragma unroll
    for (int t = 0; t < 4; ++t)
#pragma unroll
        for (int nb = 0; nb < 2; ++nb) {
            const int kl = nb * 16 + ln;
#pragma unroll
            for (int r = 0; r < 8; ++r) {
                const int i = rowBase + t * 16 + r + 8 * h;
                float x = acc[t][nb][r] + bp[i];
                unsigned short hx = f2bf(x);
                qhi[kl * QROW + i] = hx;
                qlo[kl * QROW + i] = f2bf(x - bf2f(hx));
            }
        }
    __syncthreads();

    // ------- Phase 2: out = relu(W3@M + W4@q + b3) * q ---------------------
#pragma unroll
    for (int t = 0; t < 4; ++t)
#pragma unroll
        for (int nb = 0; nb < 2; ++nb) acc[t][nb] = vzero;

    for (int jc = 0; jc < DD / JC; ++jc) {
        const int j0 = jc * JC;
        __syncthreads();
        {   // stage M[j0..j0+31][k0..k0+31] transposed
            const int p = tid * 2;
            const int j = p >> 5;
            const int k = p & 31;
            f32x2 f = *(const f32x2*)(Mb + (size_t)(j0 + j) * DD + k0 + k);
            unsigned short h0 = f2bf(f.x);
            shi[(k + 0) * SROW + j] = h0;
            slo[(k + 0) * SROW + j] = f2bf(f.x - bf2f(h0));
            unsigned short h1 = f2bf(f.y);
            shi[(k + 1) * SROW + j] = h1;
            slo[(k + 1) * SROW + j] = f2bf(f.y - bf2f(h1));
        }
        __syncthreads();
        if (jc + 1 < DD / JC) {   // prefetch next weight chunks
            __builtin_prefetch(W3Hi + (size_t)(rowBase + ln) * DD + j0 + JC, 0, 3);
            __builtin_prefetch(W4Hi + (size_t)(rowBase + ln) * DD + j0 + JC, 0, 3);
        }
        // B fragments shared across row-tiles: load once per chunk
        V16 b1hi[2], b1lo[2], b2hi[2], b2lo[2];
#pragma unroll
        for (int nb = 0; nb < 2; ++nb) {
            const int kl = nb * 16 + ln;
            loadB(shi, SROW, kl, 16 * h, b1hi[nb]);
            loadB(slo, SROW, kl, 16 * h, b1lo[nb]);
            loadB(qhi, QROW, kl, j0 + 16 * h, b2hi[nb]);
            loadB(qlo, QROW, kl, j0 + 16 * h, b2lo[nb]);
        }
#pragma unroll
        for (int t = 0; t < 4; ++t) {
            const int i0 = rowBase + t * 16;
            const size_t rowOff = (size_t)(i0 + ln) * DD + j0;
            V16 a1hi, a1lo, a2hi, a2lo;
            loadA(W3Hi, W3Lo, rowOff, h, a1hi, a1lo);
            loadA(W4Hi, W4Lo, rowOff, h, a2hi, a2lo);
#pragma unroll
            for (int nb = 0; nb < 2; ++nb) {
                v8f c = acc[t][nb];
                c = wmma_bf16(a1hi, b1hi[nb], c);
                c = wmma_bf16(a1hi, b1lo[nb], c);
                c = wmma_bf16(a1lo, b1hi[nb], c);
                c = wmma_bf16(a2hi, b2hi[nb], c);
                c = wmma_bf16(a2hi, b2lo[nb], c);
                c = wmma_bf16(a2lo, b2hi[nb], c);
                acc[t][nb] = c;
            }
        }
    }

    // epilogue: bias, relu, gate with q, store
    float* outb = out + (size_t)b * DD * DD;
#pragma unroll
    for (int t = 0; t < 4; ++t)
#pragma unroll
        for (int nb = 0; nb < 2; ++nb) {
            const int kl = nb * 16 + ln;
#pragma unroll
            for (int r = 0; r < 8; ++r) {
                const int i = rowBase + t * 16 + r + 8 * h;
                float x = acc[t][nb][r] + b3[i];
                x = fmaxf(x, 0.0f);
                float g = bf2f(qhi[kl * QROW + i]) + bf2f(qlo[kl * QROW + i]);
                outb[(size_t)i * DD + (k0 + kl)] = x * g;
            }
        }
}

extern "C" void kernel_launch(void* const* d_in, const int* in_sizes, int n_in,
                              void* d_out, int out_size, void* d_ws, size_t ws_size,
                              hipStream_t stream) {
    (void)in_sizes; (void)n_in; (void)out_size; (void)ws_size;
    const float* Q  = (const float*)d_in[1];
    const float* M  = (const float*)d_in[2];
    const float* Wp = (const float*)d_in[4];
    const float* bp = (const float*)d_in[5];
    const float* W3 = (const float*)d_in[6];
    const float* W4 = (const float*)d_in[7];
    const float* b3 = (const float*)d_in[8];
    float* out = (float*)d_out;

    // ws layout: 6 bf16 planes of DD*DD ushorts (12 MB total):
    //   [WpHi][WpLo][W3Hi][W3Lo][W4Hi][W4Lo]
    unsigned short* ws = (unsigned short*)d_ws;
    const size_t n = (size_t)DD * DD;

    weight_prep_kernel<<<dim3((unsigned)(n / 4 / 256)), dim3(256), 0, stream>>>(
        Wp, W3, W4, ws);

    coattn_fused_kernel<<<dim3(NB * (DD / KT)), dim3(512), 0, stream>>>(
        Q, M,
        ws + 0 * n, ws + 1 * n,   // Wp hi/lo
        ws + 2 * n, ws + 3 * n,   // W3 hi/lo
        ws + 4 * n, ws + 5 * n,   // W4 hi/lo
        bp, b3, out);
}
